// EquivariantMPLayer_27891517620956
// MI455X (gfx1250) — matrix operand order repaired and checked by reading, not compiled, not run on hardware
//
#include <hip/hip_runtime.h>
#include <hip/hip_bf16.h>

typedef __attribute__((ext_vector_type(16))) _Float16 v16h;
typedef __attribute__((ext_vector_type(8)))  float    v8f;
typedef __attribute__((ext_vector_type(4)))  int      v4i_t;

#define N_NODES 50000
#define N_EDGES 800000
#define DIN 64
#define DOUT 64
#define NB 16
#define K1PAD 160            /* 144 (=2*DIN+NB) padded to multiple of 32 */
#define H1 128               /* edge-MLP hidden */
#define ETILES (N_EDGES / 16)
#define NTILES (N_NODES / 16)

#define WAIT_DS() asm volatile("s_wait_dscnt 0" ::: "memory")

#if __has_builtin(__builtin_amdgcn_s_wait_asynccnt)
#define WAIT_ASYNC() __builtin_amdgcn_s_wait_asynccnt(0)
#else
#define WAIT_ASYNC() asm volatile("s_wait_asynccnt 0" ::: "memory")
#endif

// ---- async global->LDS (gfx1250): 16B per lane per op ----
// Builtin signature (from clang diagnostic): (v4i AS1*, v4i AS3*, imm off, imm cpol).
__device__ __forceinline__ unsigned lds_off32(void* p) {
  return (unsigned)(unsigned long long)(__attribute__((address_space(3))) void*)p;
}
#if __has_builtin(__builtin_amdgcn_global_load_async_to_lds_b128)
#define ASYNC_G2L_B128(g, l)                                                   \
  __builtin_amdgcn_global_load_async_to_lds_b128(                              \
      (__attribute__((address_space(1))) v4i_t*)(unsigned long long)(uintptr_t)(g), \
      (__attribute__((address_space(3))) v4i_t*)(l), 0, 0)
#else
#define ASYNC_G2L_B128(g, l)                                                   \
  asm volatile("global_load_async_to_lds_b128 %0, %1, off"                     \
               :: "v"(lds_off32((void*)(l))),                                  \
                  "v"((unsigned long long)(uintptr_t)(g))                      \
               : "memory")
#endif

// A-fragment (16x32 f16, wave32): lane holds row m=lane&15; half=lane>>4.
// h[0..15] = K[kt*32 + half*16 .. +15].
__device__ __forceinline__ v16h ld_fragA(const _Float16* p) {
  v16h r;
  *(uint4*)&r       = *(const uint4*)(p);
  *((uint4*)&r + 1) = *(const uint4*)(p + 8);
  return r;
}
// B-fragment (32x16 f16): lane holds col n=nt*16+(lane&15);
// h[0..15] = K[kt*32 + half*16 .. +15] (contiguous with transposed weights).
__device__ __forceinline__ v16h ld_fragB(const _Float16* p) {
  v16h r;
  *(uint4*)&r       = *(const uint4*)(p);
  *((uint4*)&r + 1) = *(const uint4*)(p + 8);
  return r;
}
__device__ __forceinline__ void st4h(_Float16* p, float4 v) {
  union { _Float16 h[4]; uint2 u; } t;
  t.h[0] = (_Float16)v.x; t.h[1] = (_Float16)v.y;
  t.h[2] = (_Float16)v.z; t.h[3] = (_Float16)v.w;
  *(uint2*)p = t.u;
}
__device__ __forceinline__ float silu_f(float v) { return v / (1.0f + __expf(-v)); }

#define WMMA_F16(a, b, c) \
  __builtin_amdgcn_wmma_f32_16x16x32_f16(false, (a), false, (b), (short)0, (c), false, false)

// Zero agg/cnt and convert x -> f16 (halves L2 gather traffic for the edge pass).
__global__ void prep_kernel(const float* __restrict__ x,
                            float* __restrict__ zreg, _Float16* __restrict__ xh) {
  int i = blockIdx.x * blockDim.x + threadIdx.x;
  if (i < N_NODES * DOUT + N_NODES) zreg[i] = 0.0f;
  if (i < N_NODES * DIN) xh[i] = (_Float16)x[i];
}

// ---------------- Edge message kernel ----------------
__global__ void __launch_bounds__(256) edge_msg_kernel(
    const _Float16* __restrict__ xh, const int* __restrict__ ei,
    const float* __restrict__ elen,
    const float* __restrict__ mw1, const float* __restrict__ mb1,
    const float* __restrict__ mw2, const float* __restrict__ mb2,
    float* __restrict__ agg, float* __restrict__ cnt)
{
  __shared__ __align__(16) _Float16 WT1[H1 * K1PAD];     // mw1^T: [n<128][k<160]
  __shared__ __align__(16) _Float16 WT2[DOUT * H1];      // mw2^T: [n<64][k<128]
  __shared__ __align__(16) _Float16 Abuf[8][16 * K1PAD]; // per-wave msg_in tile
  __shared__ __align__(16) _Float16 A2buf[8][16 * H1];   // per-wave hidden tile
  __shared__ int Dst[8][16];

  const int tid = threadIdx.x;
  // Stage f16 transposed weights once per block (amortized over ~78 tiles).
  for (int i = tid; i < H1 * K1PAD; i += 256) {
    int n = i / K1PAD, k = i - n * K1PAD;
    WT1[i] = (k < 2 * DIN + NB) ? (_Float16)mw1[k * H1 + n] : (_Float16)0.0f;
  }
  for (int i = tid; i < DOUT * H1; i += 256) {
    int n = i / H1, k = i - n * H1;
    WT2[i] = (_Float16)mw2[k * DOUT + n];
  }
  __syncthreads();

  const int wave = tid >> 5, lane = tid & 31;
  const int ln = lane & 15, half = lane >> 4;
  _Float16* A  = Abuf[wave];
  _Float16* A2 = A2buf[wave];

  // K-pad columns (144..159) are never overwritten: zero once.
  if (half == 0) {
#pragma unroll
    for (int b = 2 * DIN + NB; b < K1PAD; ++b) A[ln * K1PAD + b] = (_Float16)0.0f;
  }

  float bias1[8], bias2[4];
#pragma unroll
  for (int nt = 0; nt < 8; ++nt) bias1[nt] = mb1[nt * 16 + ln];
#pragma unroll
  for (int nt = 0; nt < 4; ++nt) bias2[nt] = mb2[nt * 16 + ln];

  const int gwave = blockIdx.x * 8 + wave;
  const int nwave = gridDim.x * 8;

  for (int tile = gwave; tile < ETILES; tile += nwave) {
    const int e   = tile * 16 + ln;
    const int idx = ei[half * N_EDGES + e];       // half==0: src, half==1: dst
    // Async-gather one 128B f16 row straight into LDS (no VGPR round trip).
    {
      const _Float16* gp = xh + (size_t)idx * DIN;
      _Float16* lp = &A[ln * K1PAD + half * DIN];
#pragma unroll
      for (int i = 0; i < 8; ++i) ASYNC_G2L_B128(gp + i * 8, lp + i * 8);
    }
    if (half == 0) {                              // rbf features
      const float d = elen[e];
#pragma unroll
      for (int b = 0; b < NB; ++b) {              // centers 10*b/15, 1/(2w^2)=4.5
        float t = d - (10.0f / 15.0f) * (float)b;
        A[ln * K1PAD + 2 * DIN + b] = (_Float16)__expf(-4.5f * t * t);
      }
    } else {
      Dst[wave][ln] = idx;
      atomicAdd(&cnt[idx], 1.0f);
    }
    WAIT_ASYNC();
    WAIT_DS();

    // ---- GEMM1: (16x160) x (160x128), B-frags batched 4-wide per K-step ----
    v8f acc[8];
#pragma unroll
    for (int nt = 0; nt < 8; ++nt) {
      v8f c;
#pragma unroll
      for (int r = 0; r < 8; ++r) c[r] = bias1[nt];
      acc[nt] = c;
    }
#pragma unroll
    for (int kt = 0; kt < 5; ++kt) {
      const v16h ak = ld_fragA(&A[ln * K1PAD + kt * 32 + half * 16]);
#pragma unroll
      for (int ng = 0; ng < 2; ++ng) {
        const int nb = ng * 4;
        v16h b0 = ld_fragB(&WT1[((nb + 0) * 16 + ln) * K1PAD + kt * 32 + half * 16]);
        v16h b1 = ld_fragB(&WT1[((nb + 1) * 16 + ln) * K1PAD + kt * 32 + half * 16]);
        v16h b2 = ld_fragB(&WT1[((nb + 2) * 16 + ln) * K1PAD + kt * 32 + half * 16]);
        v16h b3 = ld_fragB(&WT1[((nb + 3) * 16 + ln) * K1PAD + kt * 32 + half * 16]);
        acc[nb + 0] = WMMA_F16(ak, b0, acc[nb + 0]);
        acc[nb + 1] = WMMA_F16(ak, b1, acc[nb + 1]);
        acc[nb + 2] = WMMA_F16(ak, b2, acc[nb + 2]);
        acc[nb + 3] = WMMA_F16(ak, b3, acc[nb + 3]);
      }
    }
    // SiLU -> A2 (f16). C/D layout: element (m = r+8*half, n = nt*16+ln).
#pragma unroll
    for (int nt = 0; nt < 8; ++nt) {
#pragma unroll
      for (int r = 0; r < 8; ++r)
        A2[(r + 8 * half) * H1 + nt * 16 + ln] = (_Float16)silu_f(acc[nt][r]);
    }
    WAIT_DS();

    // ---- GEMM2: (16x128) x (128x64) ----
    v8f acc2[4];
#pragma unroll
    for (int nt = 0; nt < 4; ++nt) {
      v8f c;
#pragma unroll
      for (int r = 0; r < 8; ++r) c[r] = bias2[nt];
      acc2[nt] = c;
    }
#pragma unroll
    for (int kt = 0; kt < 4; ++kt) {
      const v16h ak = ld_fragA(&A2[ln * H1 + kt * 32 + half * 16]);
      v16h b0 = ld_fragB(&WT2[(0 * 16 + ln) * H1 + kt * 32 + half * 16]);
      v16h b1 = ld_fragB(&WT2[(1 * 16 + ln) * H1 + kt * 32 + half * 16]);
      v16h b2 = ld_fragB(&WT2[(2 * 16 + ln) * H1 + kt * 32 + half * 16]);
      v16h b3 = ld_fragB(&WT2[(3 * 16 + ln) * H1 + kt * 32 + half * 16]);
      acc2[0] = WMMA_F16(ak, b0, acc2[0]);
      acc2[1] = WMMA_F16(ak, b1, acc2[1]);
      acc2[2] = WMMA_F16(ak, b2, acc2[2]);
      acc2[3] = WMMA_F16(ak, b3, acc2[3]);
    }
    // Scatter msg into agg[dst] (L2-resident float atomics).
    int nodes[8];
#pragma unroll
    for (int r = 0; r < 8; ++r) nodes[r] = Dst[wave][r + 8 * half];
#pragma unroll
    for (int nt = 0; nt < 4; ++nt) {
#pragma unroll
      for (int r = 0; r < 8; ++r)
        atomicAdd(&agg[(size_t)nodes[r] * DOUT + nt * 16 + ln], acc2[nt][r]);
    }
  }
}

// ---------------- Node update + LayerNorm kernel ----------------
__global__ void __launch_bounds__(256) node_update_kernel(
    const float* __restrict__ x, const float* __restrict__ agg,
    const float* __restrict__ cnt,
    const float* __restrict__ uw1, const float* __restrict__ ub1,
    const float* __restrict__ uw2, const float* __restrict__ ub2,
    const float* __restrict__ lng, const float* __restrict__ lnb,
    float* __restrict__ out)
{
  __shared__ __align__(16) _Float16 WU1[DOUT * 128];     // uw1^T: [n<64][k<128]
  __shared__ __align__(16) _Float16 WU2[DOUT * DOUT];    // uw2^T: [n<64][k<64]
  __shared__ __align__(16) _Float16 Abuf[8][16 * 128];
  __shared__ __align__(16) _Float16 A2buf[8][16 * 64];
  __shared__ __align__(16) float    Obuf[8][16 * 64];

  const int tid = threadIdx.x;
  for (int i = tid; i < DOUT * 128; i += 256) {
    int n = i / 128, k = i - n * 128;
    WU1[i] = (_Float16)uw1[k * DOUT + n];
  }
  for (int i = tid; i < DOUT * DOUT; i += 256) {
    int n = i / DOUT, k = i - n * DOUT;
    WU2[i] = (_Float16)uw2[k * DOUT + n];
  }
  __syncthreads();

  const int wave = tid >> 5, lane = tid & 31;
  const int ln = lane & 15, half = lane >> 4;
  _Float16* A  = Abuf[wave];
  _Float16* A2 = A2buf[wave];
  float*    Ob = Obuf[wave];

  float b1[4], b2[4];
#pragma unroll
  for (int nt = 0; nt < 4; ++nt) { b1[nt] = ub1[nt * 16 + ln]; b2[nt] = ub2[nt * 16 + ln]; }

  const int gwave = blockIdx.x * 8 + wave;
  const int nwave = gridDim.x * 8;

  for (int tile = gwave; tile < NTILES; tile += nwave) {
    const int node = tile * 16 + ln;
    if (half == 0) {                       // upd_in[:, 0:64] = x
      const float4* xr = (const float4*)(x + (size_t)node * DIN);
#pragma unroll
      for (int i = 0; i < 16; ++i) st4h(&A[ln * 128 + i * 4], xr[i]);
    } else {                               // upd_in[:, 64:128] = agg / max(cnt,1)
      const float inv = 1.0f / fmaxf(cnt[node], 1.0f);
      const float4* ar = (const float4*)(agg + (size_t)node * DOUT);
#pragma unroll
      for (int i = 0; i < 16; ++i) {
        float4 v = ar[i];
        v.x *= inv; v.y *= inv; v.z *= inv; v.w *= inv;
        st4h(&A[ln * 128 + DIN + i * 4], v);
      }
    }
    WAIT_DS();

    // GEMM1: (16x128) x (128x64)
    v8f acc[4];
#pragma unroll
    for (int nt = 0; nt < 4; ++nt) {
      v8f c;
#pragma unroll
      for (int r = 0; r < 8; ++r) c[r] = b1[nt];
      acc[nt] = c;
    }
#pragma unroll
    for (int kt = 0; kt < 4; ++kt) {
      const v16h ak = ld_fragA(&A[ln * 128 + kt * 32 + half * 16]);
      v16h b0 = ld_fragB(&WU1[(0 * 16 + ln) * 128 + kt * 32 + half * 16]);
      v16h b1f = ld_fragB(&WU1[(1 * 16 + ln) * 128 + kt * 32 + half * 16]);
      v16h b2f = ld_fragB(&WU1[(2 * 16 + ln) * 128 + kt * 32 + half * 16]);
      v16h b3f = ld_fragB(&WU1[(3 * 16 + ln) * 128 + kt * 32 + half * 16]);
      acc[0] = WMMA_F16(ak, b0, acc[0]);
      acc[1] = WMMA_F16(ak, b1f, acc[1]);
      acc[2] = WMMA_F16(ak, b2f, acc[2]);
      acc[3] = WMMA_F16(ak, b3f, acc[3]);
    }
#pragma unroll
    for (int nt = 0; nt < 4; ++nt) {
#pragma unroll
      for (int r = 0; r < 8; ++r)
        A2[(r + 8 * half) * DOUT + nt * 16 + ln] = (_Float16)silu_f(acc[nt][r]);
    }
    WAIT_DS();

    // GEMM2: (16x64) x (64x64)
    v8f acc2[4];
#pragma unroll
    for (int nt = 0; nt < 4; ++nt) {
      v8f c;
#pragma unroll
      for (int r = 0; r < 8; ++r) c[r] = b2[nt];
      acc2[nt] = c;
    }
#pragma unroll
    for (int kt = 0; kt < 2; ++kt) {
      const v16h ak = ld_fragA(&A2[ln * DOUT + kt * 32 + half * 16]);
      v16h b0 = ld_fragB(&WU2[(0 * 16 + ln) * DOUT + kt * 32 + half * 16]);
      v16h b1f = ld_fragB(&WU2[(1 * 16 + ln) * DOUT + kt * 32 + half * 16]);
      v16h b2f = ld_fragB(&WU2[(2 * 16 + ln) * DOUT + kt * 32 + half * 16]);
      v16h b3f = ld_fragB(&WU2[(3 * 16 + ln) * DOUT + kt * 32 + half * 16]);
      acc2[0] = WMMA_F16(ak, b0, acc2[0]);
      acc2[1] = WMMA_F16(ak, b1f, acc2[1]);
      acc2[2] = WMMA_F16(ak, b2f, acc2[2]);
      acc2[3] = WMMA_F16(ak, b3f, acc2[3]);
    }
    // Stage pre-LN rows, then per-row LayerNorm by lanes 0-15.
#pragma unroll
    for (int nt = 0; nt < 4; ++nt) {
#pragma unroll
      for (int r = 0; r < 8; ++r)
        Ob[(r + 8 * half) * DOUT + nt * 16 + ln] = acc2[nt][r];
    }
    WAIT_DS();

    if (half == 0) {
      const float* row = &Ob[ln * DOUT];
      float mu = 0.0f;
#pragma unroll
      for (int j = 0; j < DOUT; ++j) mu += row[j];
      mu *= (1.0f / DOUT);
      float var = 0.0f;
#pragma unroll
      for (int j = 0; j < DOUT; ++j) { float d = row[j] - mu; var += d * d; }
      var *= (1.0f / DOUT);
      const float inv = rsqrtf(var + 1e-5f);
      float* op = out + (size_t)node * DOUT;
#pragma unroll
      for (int j = 0; j < DOUT; ++j)
        op[j] = (row[j] - mu) * inv * lng[j] + lnb[j];
    }
    WAIT_DS();   // keep Obuf reads ahead of next iteration's overwrites
  }
}

extern "C" void kernel_launch(void* const* d_in, const int* in_sizes, int n_in,
                              void* d_out, int out_size, void* d_ws, size_t ws_size,
                              hipStream_t stream) {
  const float* x    = (const float*)d_in[0];
  const int*   ei   = (const int*)d_in[1];     // edge_index (2,E) -> int32
  /* d_in[2] edge_vec: unused by the reference math */
  const float* elen = (const float*)d_in[3];
  const float* mw1  = (const float*)d_in[4];
  const float* mb1  = (const float*)d_in[5];
  const float* mw2  = (const float*)d_in[6];
  const float* mb2  = (const float*)d_in[7];
  const float* uw1  = (const float*)d_in[8];
  const float* ub1  = (const float*)d_in[9];
  const float* uw2  = (const float*)d_in[10];
  const float* ub2  = (const float*)d_in[11];
  const float* lng  = (const float*)d_in[12];
  const float* lnb  = (const float*)d_in[13];
  float* out = (float*)d_out;

  float*     agg = (float*)d_ws;                      // N*64 f32
  float*     cnt = agg + (size_t)N_NODES * DOUT;      // N f32
  _Float16*  xh  = (_Float16*)(cnt + N_NODES);        // N*64 f16 (16B-aligned offset)

  const int ptot = N_NODES * DOUT + N_NODES;          // covers both prep ranges
  prep_kernel<<<(ptot + 255) / 256, 256, 0, stream>>>(x, agg, xh);
  edge_msg_kernel<<<640, 256, 0, stream>>>(xh, ei, elen, mw1, mb1, mw2, mb2, agg, cnt);
  node_update_kernel<<<400, 256, 0, stream>>>(x, agg, cnt, uw1, ub1, uw2, ub2, lng, lnb, out);
}
